// TopKMoE_77429670413047
// MI455X (gfx1250) — matrix operand order repaired and checked by reading, not compiled, not run on hardware
//
#include <hip/hip_runtime.h>
#include <hip/hip_bf16.h>
#include <math.h>

#define N_TOK 16384
#define DIM   512
#define HID   2048
#define NE    8
#define MT    32            // tokens per FFN block (2 WMMA M-tiles)

typedef __attribute__((ext_vector_type(16))) _Float16 v16h;
typedef __attribute__((ext_vector_type(8)))  _Float16 v8h;
typedef __attribute__((ext_vector_type(8)))  float    v8f;
typedef __attribute__((ext_vector_type(4)))  int      v4i;

#define XS_STRIDE 520   // 512 + 8 halfs -> 1040B rows (16B aligned, bank-staggered)
#define HS_STRIDE 136   // 128 + 8 halfs -> 272B rows

#if defined(__has_builtin)
#if __has_builtin(__builtin_amdgcn_global_load_async_to_lds_b128)
#define HAVE_ASYNC_LDS 1
#endif
#if __has_builtin(__builtin_amdgcn_s_wait_asynccnt)
#define HAVE_WAIT_ASYNC 1
#endif
#endif
#ifndef HAVE_ASYNC_LDS
#define HAVE_ASYNC_LDS 0
#endif
#ifndef HAVE_WAIT_ASYNC
#define HAVE_WAIT_ASYNC 0
#endif

// ---------------------------------------------------------------------------
// Kernel 0: f32->f16 conversions into workspace; zero out/counts.
//   W1 [E][D][H] -> W1t [E][H][D]  (B-operand of GEMM1: contiguous K runs)
//   W2 [E][H][D] -> W2t [E][D][H]  (B-operand of GEMM2: contiguous K runs)
//   x  [N][D]    -> xh  [N][D]     (enables raw async b128 gather to LDS)
// ---------------------------------------------------------------------------
__global__ __launch_bounds__(256)
void moe_prep_kernel(const float* __restrict__ x,
                     const float* __restrict__ W1, const float* __restrict__ W2,
                     _Float16* __restrict__ W1t, _Float16* __restrict__ W2t,
                     _Float16* __restrict__ xh,
                     float* __restrict__ out, int* __restrict__ counts)
{
    const int total = NE * DIM * HID;      // == N_TOK * DIM == 8388608
    int gid = blockIdx.x * blockDim.x + threadIdx.x;
    if (gid < NE) counts[gid] = 0;
    for (int idx = gid; idx < total; idx += gridDim.x * blockDim.x) {
        {   // W1 transpose
            int e = idx / (DIM * HID);
            int r = idx - e * (DIM * HID);
            int d = r / HID;
            int h = r - d * HID;
            W1t[((size_t)e * HID + h) * DIM + d] = (_Float16)W1[idx];
        }
        {   // W2 transpose
            int e = idx / (HID * DIM);
            int r = idx - e * (HID * DIM);
            int h = r / DIM;
            int d = r - h * DIM;
            W2t[((size_t)e * DIM + d) * HID + h] = (_Float16)W2[idx];
        }
        xh[idx]  = (_Float16)x[idx];
        out[idx] = 0.0f;   // zero output (epilogue uses atomic accumulation)
    }
}

// ---------------------------------------------------------------------------
// Kernel 1: router. One wave per token: logits = x_row @ Wr, top-2 softmax,
// dense gate row, append token to per-expert lists.
// ---------------------------------------------------------------------------
__global__ __launch_bounds__(128)
void moe_router_kernel(const float* __restrict__ x, const float* __restrict__ Wr,
                       float* __restrict__ gates, int* __restrict__ counts,
                       int* __restrict__ lists)
{
    const int wave = threadIdx.x >> 5;
    const int lane = threadIdx.x & 31;
    const int n = blockIdx.x * 4 + wave;
    if (n >= N_TOK) return;

    float acc[NE];
#pragma unroll
    for (int e = 0; e < NE; ++e) acc[e] = 0.0f;

    const float* xr = x + (size_t)n * DIM;
    for (int d = lane; d < DIM; d += 32) {
        float xv = xr[d];
#pragma unroll
        for (int e = 0; e < NE; ++e) acc[e] += xv * Wr[d * NE + e];
    }
    float lg[NE];
#pragma unroll
    for (int e = 0; e < NE; ++e) {
        float v = acc[e];
        for (int off = 16; off > 0; off >>= 1) v += __shfl_down(v, off, 32);
        lg[e] = v;   // valid on lane 0
    }
    if (lane == 0) {
        int bi = 0;
        float best = lg[0];
#pragma unroll
        for (int e = 1; e < NE; ++e) if (lg[e] > best) { best = lg[e]; bi = e; }
        int si = (bi == 0) ? 1 : 0;
        float second = lg[si];
#pragma unroll
        for (int e = 0; e < NE; ++e)
            if (e != bi && lg[e] > second) { second = lg[e]; si = e; }
        float w1 = expf(second - best);
        float s  = 1.0f + w1;
        float g0 = 1.0f / s;
        float g1 = w1 / s;
#pragma unroll
        for (int e = 0; e < NE; ++e)
            gates[(size_t)n * NE + e] = (e == bi) ? g0 : ((e == si) ? g1 : 0.0f);
        int p0 = atomicAdd(&counts[bi], 1);
        lists[(size_t)bi * N_TOK + p0] = n;
        int p1 = atomicAdd(&counts[si], 1);
        lists[(size_t)si * N_TOK + p1] = n;
    }
}

// ---------------------------------------------------------------------------
// Kernel 2: expert FFN over gathered 32-token tiles, f16 WMMA, f32 accum.
// grid = (N_TOK/32, NE), block = 256 (8 waves).
//   GEMM1: wave w -> hdn cols [hc+16w, +16), both 16-row M-tiles (B reused).
//   GEMM2: wave w -> y cols   [64w, +64),  both M-tiles (B reused).
// ---------------------------------------------------------------------------
__global__ __launch_bounds__(256)
void moe_ffn_kernel(const _Float16* __restrict__ xh,
                    const _Float16* __restrict__ W1t, const _Float16* __restrict__ W2t,
                    const float* __restrict__ b1, const float* __restrict__ b2,
                    const float* __restrict__ gates, const int* __restrict__ counts,
                    const int* __restrict__ lists, float* __restrict__ out)
{
    const int e = blockIdx.y;
    const int cnt = counts[e];
    const int tilebase = blockIdx.x * MT;
    if (tilebase >= cnt) return;   // uniform across block

    __shared__ _Float16 xs[MT * XS_STRIDE];  // gathered x tile (f16), padded rows
    __shared__ _Float16 hs[MT * HS_STRIDE];  // gelu(hdn) chunk [32 x 128], padded
    __shared__ int   toks[MT];
    __shared__ float gts[MT];

    const int tid  = threadIdx.x;
    const int wave = tid >> 5;
    const int lane = tid & 31;
    const int g    = lane >> 4;    // WMMA lane half-group
    const int nl   = lane & 15;    // n / m within 16

    if (tid < MT) {
        int slot = tilebase + tid;
        int t = (slot < cnt) ? lists[(size_t)e * N_TOK + slot]
                             : lists[(size_t)e * N_TOK + tilebase];
        toks[tid] = t;
        gts[tid]  = (slot < cnt) ? gates[(size_t)t * NE + e] : 0.0f;
    }
    __syncthreads();

    // Stage gathered rows: 32 rows x 64 b128-chunks = 2048 chunks.
    for (int c = tid; c < MT * 64; c += 256) {
        int row = c >> 6;
        int col = c & 63;
        const _Float16* gsrc = xh + (size_t)toks[row] * DIM + col * 8;
        _Float16*       ldst = &xs[row * XS_STRIDE + col * 8];
#if HAVE_ASYNC_LDS
        __builtin_amdgcn_global_load_async_to_lds_b128(
            (v4i*)gsrc, (v4i*)ldst, 0, 0);
#else
        *(v8h*)ldst = *(const v8h*)gsrc;
#endif
    }
#if HAVE_ASYNC_LDS
#if HAVE_WAIT_ASYNC
    __builtin_amdgcn_s_wait_asynccnt(0);
#else
    asm volatile("s_wait_asynccnt 0" ::: "memory");
#endif
#endif
    __syncthreads();

    const _Float16* W1e = W1t + (size_t)e * HID * DIM;   // [H][D]
    const _Float16* W2e = W2t + (size_t)e * DIM * HID;   // [D][H]
    const float*    b1e = b1 + e * HID;
    const float*    b2e = b2 + e * DIM;

    v8f acc[2][4];
#pragma unroll
    for (int mt = 0; mt < 2; ++mt)
#pragma unroll
        for (int t = 0; t < 4; ++t)
#pragma unroll
            for (int q = 0; q < 8; ++q) acc[mt][t][q] = 0.0f;

    for (int hc = 0; hc < HID; hc += 128) {
        // ---- GEMM1: two hdn tiles [16x16] at cols hc + 16*wave, K = 512 ----
        v8f h0, h1;
#pragma unroll
        for (int q = 0; q < 8; ++q) { h0[q] = 0.0f; h1[q] = 0.0f; }

        const int hcol = hc + wave * 16 + nl;
        const _Float16* Bb = W1e + (size_t)hcol * DIM;   // contiguous over d
        if (hc + 128 < HID)  // prefetch next chunk's W1 row
            __builtin_prefetch(W1e + (size_t)(hcol + 128) * DIM, 0, 0);
#pragma unroll
        for (int kk = 0; kk < 16; ++kk) {
            union { v16h v; v8h h[2]; } A0, A1, B;
            B.h[0]  = *(const v8h*)(Bb + kk * 32 + g * 8);
            B.h[1]  = *(const v8h*)(Bb + kk * 32 + 16 + g * 8);
            A0.h[0] = *(const v8h*)(&xs[nl * XS_STRIDE + kk * 32 + g * 8]);
            A0.h[1] = *(const v8h*)(&xs[nl * XS_STRIDE + kk * 32 + 16 + g * 8]);
            A1.h[0] = *(const v8h*)(&xs[(nl + 16) * XS_STRIDE + kk * 32 + g * 8]);
            A1.h[1] = *(const v8h*)(&xs[(nl + 16) * XS_STRIDE + kk * 32 + 16 + g * 8]);
            h0 = __builtin_amdgcn_wmma_f32_16x16x32_f16(
                false, A0.v, false, B.v, (short)0, h0, false, false);
            h1 = __builtin_amdgcn_wmma_f32_16x16x32_f16(
                false, A1.v, false, B.v, (short)0, h1, false, false);
        }

        float b1v = b1e[hcol];
        __syncthreads();   // previous-iter hs readers done
#pragma unroll
        for (int i = 0; i < 8; ++i) {
            int m = i + g * 8;
            float v0  = h0[i] + b1v;
            float v1  = h1[i] + b1v;
            float ge0 = 0.5f * v0 * (1.0f + erff(v0 * 0.70710678118654752f));
            float ge1 = 0.5f * v1 * (1.0f + erff(v1 * 0.70710678118654752f));
            hs[m * HS_STRIDE + wave * 16 + nl]        = (_Float16)ge0;
            hs[(m + 16) * HS_STRIDE + wave * 16 + nl] = (_Float16)ge1;
        }
        __syncthreads();

        // ---- GEMM2: y[32 x 64] += hs[32 x 128] @ W2t chunk ----
#pragma unroll
        for (int kk = 0; kk < 4; ++kk) {
            union { v16h v; v8h h[2]; } A0, A1;
            A0.h[0] = *(const v8h*)(&hs[nl * HS_STRIDE + kk * 32 + g * 8]);
            A0.h[1] = *(const v8h*)(&hs[nl * HS_STRIDE + kk * 32 + 16 + g * 8]);
            A1.h[0] = *(const v8h*)(&hs[(nl + 16) * HS_STRIDE + kk * 32 + g * 8]);
            A1.h[1] = *(const v8h*)(&hs[(nl + 16) * HS_STRIDE + kk * 32 + 16 + g * 8]);
#pragma unroll
            for (int t = 0; t < 4; ++t) {
                int ncol = wave * 64 + t * 16 + nl;
                const _Float16* B2 = W2e + (size_t)ncol * HID + hc;
                union { v16h v; v8h h[2]; } B;
                B.h[0] = *(const v8h*)(B2 + kk * 32 + g * 8);
                B.h[1] = *(const v8h*)(B2 + kk * 32 + 16 + g * 8);
                if (kk == 0 && hc + 128 < HID)   // prefetch next chunk's W2 row
                    __builtin_prefetch(B2 + 128, 0, 0);
                acc[0][t] = __builtin_amdgcn_wmma_f32_16x16x32_f16(
                    false, A0.v, false, B.v, (short)0, acc[0][t], false, false);
                acc[1][t] = __builtin_amdgcn_wmma_f32_16x16x32_f16(
                    false, A1.v, false, B.v, (short)0, acc[1][t], false, false);
            }
        }
    }

    // ---- Epilogue: out[tok][n] += gate * (y + b2)  (atomic: 2 experts/token)
#pragma unroll
    for (int t = 0; t < 4; ++t) {
        int ncol = wave * 64 + t * 16 + nl;
        float bb = b2e[ncol];
#pragma unroll
        for (int mt = 0; mt < 2; ++mt)
#pragma unroll
            for (int i = 0; i < 8; ++i) {
                int m = i + g * 8 + mt * 16;
                float val = (acc[mt][t][i] + bb) * gts[m];
                atomicAdd(out + (size_t)toks[m] * DIM + ncol, val);
            }
    }
}

// ---------------------------------------------------------------------------
extern "C" void kernel_launch(void* const* d_in, const int* in_sizes, int n_in,
                              void* d_out, int out_size, void* d_ws, size_t ws_size,
                              hipStream_t stream)
{
    const float* x  = (const float*)d_in[0];
    const float* Wr = (const float*)d_in[1];
    const float* W1 = (const float*)d_in[2];
    const float* b1 = (const float*)d_in[3];
    const float* W2 = (const float*)d_in[4];
    const float* b2 = (const float*)d_in[5];
    float* out = (float*)d_out;

    char* ws = (char*)d_ws;
    float*    gates  = (float*)ws;                                // 512KB
    int*      counts = (int*)(ws + (size_t)N_TOK * NE * 4);       // 8 ints (pad 256)
    int*      lists  = (int*)(ws + (size_t)N_TOK * NE * 4 + 256); // 512KB
    _Float16* W1t    = (_Float16*)(ws + (size_t)N_TOK * NE * 8 + 256); // 16MB
    _Float16* W2t    = W1t + (size_t)NE * HID * DIM;                   // 16MB
    _Float16* xh     = W2t + (size_t)NE * HID * DIM;                   // 16MB

    moe_prep_kernel<<<4096, 256, 0, stream>>>(x, W1, W2, W1t, W2t, xh, out, counts);
    moe_router_kernel<<<N_TOK / 4, 128, 0, stream>>>(x, Wr, gates, counts, lists);
    dim3 grid(N_TOK / MT, NE);
    moe_ffn_kernel<<<grid, 256, 0, stream>>>(xh, W1t, W2t, b1, b2, gates,
                                             counts, lists, out);
}
